// SimpleSNNBrain_25288767438881
// MI455X (gfx1250) — matrix operand order repaired and checked by reading, not compiled, not run on hardware
//
#include <hip/hip_runtime.h>
#include <hip/hip_bf16.h>
#include <math.h>

// Problem constants (from reference)
#define N_VIS   8192
#define N_ASSOC 4096
#define N_MOTOR 1024
#define IN_DIM  4096
#define T_STEPS 16
#define DECAY_M 0.95f        // 1 - 1/20
#define DECAY_E 0.998f       // 1 - 1/500
#define V_TH    1.0f

typedef __attribute__((ext_vector_type(16))) __bf16 v16bf;
typedef __attribute__((ext_vector_type(8)))  float  v8f;

__device__ __forceinline__ float clampf(float x, float lo, float hi) {
    return fminf(fmaxf(x, lo), hi);
}

// Build a bf16 WMMA fragment from two contiguous 8-float runs (4x b128 loads).
__device__ __forceinline__ v16bf load_frag_bf16(const float* __restrict__ p0,
                                                const float* __restrict__ p1)
{
    const float4 x0 = *(const float4*)(p0);
    const float4 x1 = *(const float4*)(p0 + 4);
    const float4 y0 = *(const float4*)(p1);
    const float4 y1 = *(const float4*)(p1 + 4);
    v16bf f;
    f[0]  = (__bf16)x0.x; f[1]  = (__bf16)x0.y; f[2]  = (__bf16)x0.z; f[3]  = (__bf16)x0.w;
    f[4]  = (__bf16)x1.x; f[5]  = (__bf16)x1.y; f[6]  = (__bf16)x1.z; f[7]  = (__bf16)x1.w;
    f[8]  = (__bf16)y0.x; f[9]  = (__bf16)y0.y; f[10] = (__bf16)y0.z; f[11] = (__bf16)y0.w;
    f[12] = (__bf16)y1.x; f[13] = (__bf16)y1.y; f[14] = (__bf16)y1.z; f[15] = (__bf16)y1.w;
    return f;
}

// ---------------------------------------------------------------------------
// Encoder: H = X * Wenc^T  (X:[16,4096], Wenc:[8192,4096]) via WMMA bf16.
// One wave per 16(M=T) x 16(N) output tile; K-loop unrolled x2 (64 K/iter).
// Epilogue: sigmoid(h + b) * 0.3 threshold vs noise -> in_spk [16, 8192].
// ---------------------------------------------------------------------------
__global__ void __launch_bounds__(256)
snn_encoder_wmma(const float* __restrict__ X,
                 const float* __restrict__ Wenc,
                 const float* __restrict__ benc,
                 const float* __restrict__ noise,
                 float* __restrict__ in_spk)
{
    const int lane = threadIdx.x & 31;
    const int wave = threadIdx.x >> 5;
    const int tile = blockIdx.x * (blockDim.x >> 5) + wave;   // 0..511
    const int n0   = tile * 16;
    const int m    = lane & 15;        // A row (timestep) / B column within tile
    const bool hi  = lane >= 16;
    const int h8   = hi ? 8 : 0;

    v8f c = {};
    const float* arow  = X + (size_t)m * IN_DIM;               // X row = timestep m
    const float* browp = Wenc + (size_t)(n0 + m) * IN_DIM + (hi ? 8 : 0);

    for (int k0 = 0; k0 < IN_DIM; k0 += 64) {
        // A 16x32 bf16 layout (ISA 7.12.2): elements 0..7 -> K = k0+h8..+7,
        // elements 8..15 -> K = k0+16+h8..+7 (contiguous runs of 8).
        // B 32x16 bf16 layout: lane%16 = N, same K order as A
        // (elements 0..7 -> K = h8..+7, elements 8..15 -> K = 16+h8..+7).
        v16bf a0 = load_frag_bf16(arow + k0 + h8,       arow + k0 + 16 + h8);
        v16bf b0 = load_frag_bf16(browp + k0,           browp + k0 + 16);
        v16bf a1 = load_frag_bf16(arow + k0 + 32 + h8,  arow + k0 + 48 + h8);
        v16bf b1 = load_frag_bf16(browp + k0 + 32,      browp + k0 + 48);
        c = __builtin_amdgcn_wmma_f32_16x16x32_bf16(false, a0, false, b0,
                                                    (short)0, c, false, false);
        c = __builtin_amdgcn_wmma_f32_16x16x32_bf16(false, a1, false, b1,
                                                    (short)0, c, false, false);
    }

    // C layout: element r -> (M = r + 8*hi, N = lane%16)
    const int n = n0 + m;
    const float bias = benc[n];
    #pragma unroll
    for (int r = 0; r < 8; ++r) {
        const int t = r + h8;
        const float h = c[r] + bias;
        // fast sigmoid: v_exp_f32 + v_rcp_f32 (feeds a Bernoulli threshold;
        // ~1 ulp rcp is ample — avoids the IEEE div_scale/div_fmas chain)
        const float rate = __builtin_amdgcn_rcpf(1.0f + __expf(-h));
        const float u = noise[(size_t)t * N_VIS + n];
        in_spk[(size_t)t * N_VIS + n] = (u < rate * 0.3f) ? 1.0f : 0.0f;
    }
}

// ---------------------------------------------------------------------------
// Visual layer LIF: vV = vV*decay + in_spk; spike; reset.
// ---------------------------------------------------------------------------
__global__ void snn_vis_step(float* __restrict__ vV,
                             const float* __restrict__ in_spk_t,
                             float* __restrict__ v_spk)
{
    const int i = blockIdx.x * blockDim.x + threadIdx.x;
    float v = vV[i] * DECAY_M + in_spk_t[i];
    const float s = (v > V_TH) ? 1.0f : 0.0f;
    vV[i] = v * (1.0f - s);
    v_spk[i] = s;
}

// ---------------------------------------------------------------------------
// LIF matvec layer: h = 0.1 * W[row,:] . spk_in ; membrane update; spike.
// One wave per output row, float4 coalesced row streaming + prefetch.
// ---------------------------------------------------------------------------
__global__ void __launch_bounds__(256)
snn_lif_matvec(const float* __restrict__ W,
               const float* __restrict__ spk_in,
               float* __restrict__ V,
               float* __restrict__ spk_out,
               int C)
{
    const int row  = blockIdx.x * (blockDim.x >> 5) + (threadIdx.x >> 5);
    const int lane = threadIdx.x & 31;
    const float* wr = W + (size_t)row * C;

    float acc = 0.0f;
    for (int c = lane * 4; c < C; c += 32 * 4) {
        __builtin_prefetch(wr + c + 1024, 0, 0);   // global_prefetch_b8
        const float4 w = *(const float4*)(wr + c);
        const float4 s = *(const float4*)(spk_in + c);
        acc += w.x * s.x + w.y * s.y + w.z * s.z + w.w * s.w;
    }
    #pragma unroll
    for (int off = 16; off > 0; off >>= 1)
        acc += __shfl_xor(acc, off, 32);

    if (lane == 0) {
        float v = V[row] * DECAY_M + acc * 0.1f;
        const float s = (v > V_TH) ? 1.0f : 0.0f;
        V[row] = v * (1.0f - s);
        spk_out[row] = s;
    }
}

// ---------------------------------------------------------------------------
// Fused eligibility-trace + dopamine-gated weight update.
// E and W both stored [R, C] (eligibility kept transposed vs reference so the
// rank-1 update + weight RMW is one fully coalesced float4 stream).
//   E = E*decay_e + rowspk[r]*colspk[c]*0.01
//   if |d|>0.1:  W = clip(W + clip(d*E*0.1, ±0.01), 0, 1);  E *= 0.5
// ---------------------------------------------------------------------------
__global__ void __launch_bounds__(256)
snn_elig_update(float* __restrict__ E,
                float* __restrict__ W,
                const float* __restrict__ rowspk,
                const float* __restrict__ colspk,
                const float* __restrict__ dopa,
                int t, int C)
{
    const size_t idx = ((size_t)blockIdx.x * blockDim.x + threadIdx.x) * 4;
    const int row = (int)(idx / (size_t)C);
    const int col = (int)(idx % (size_t)C);

    const float d = dopa[t];
    const bool gate = fabsf(d) > 0.1f;

    const float rs = rowspk[row] * 0.01f;
    const float4 cs = *(const float4*)(colspk + col);
    float4 e = *(float4*)(E + idx);

    e.x = e.x * DECAY_E + rs * cs.x;
    e.y = e.y * DECAY_E + rs * cs.y;
    e.z = e.z * DECAY_E + rs * cs.z;
    e.w = e.w * DECAY_E + rs * cs.w;

    if (gate) {
        float4 w = *(float4*)(W + idx);
        w.x = clampf(w.x + clampf(d * e.x * 0.1f, -0.01f, 0.01f), 0.0f, 1.0f);
        w.y = clampf(w.y + clampf(d * e.y * 0.1f, -0.01f, 0.01f), 0.0f, 1.0f);
        w.z = clampf(w.z + clampf(d * e.z * 0.1f, -0.01f, 0.01f), 0.0f, 1.0f);
        w.w = clampf(w.w + clampf(d * e.w * 0.1f, -0.01f, 0.01f), 0.0f, 1.0f);
        *(float4*)(W + idx) = w;
        e.x *= 0.5f; e.y *= 0.5f; e.z *= 0.5f; e.w *= 0.5f;
    }
    *(float4*)(E + idx) = e;
}

// ---------------------------------------------------------------------------
extern "C" void kernel_launch(void* const* d_in, const int* in_sizes, int n_in,
                              void* d_out, int out_size, void* d_ws, size_t ws_size,
                              hipStream_t stream)
{
    (void)in_sizes; (void)n_in; (void)out_size; (void)ws_size;

    const float* visual = (const float*)d_in[0];   // [16, 4096]
    const float* noise  = (const float*)d_in[1];   // [16, 8192]
    const float* dopa   = (const float*)d_in[2];   // [16]
    const float* Wenc   = (const float*)d_in[3];   // [8192, 4096]
    const float* benc   = (const float*)d_in[4];   // [8192]
    const float* Wva_in = (const float*)d_in[5];   // [4096, 8192]
    const float* Wam_in = (const float*)d_in[6];   // [1024, 4096]
    float* out = (float*)d_out;                    // [16, 1024]

    // Workspace partition (floats). States first so one memset clears them.
    float* ws = (float*)d_ws;
    size_t o = 0;
    float* E_va  = ws + o; o += (size_t)N_ASSOC * N_VIS;   // [assoc, vis]  (eVA^T)
    float* E_am  = ws + o; o += (size_t)N_MOTOR * N_ASSOC; // [motor, assoc](eAM^T)
    float* vV    = ws + o; o += N_VIS;
    float* vA    = ws + o; o += N_ASSOC;
    float* vM    = ws + o; o += N_MOTOR;
    const size_t zero_floats = o;                 // everything above starts at 0
    float* v_spk = ws + o; o += N_VIS;
    float* a_spk = ws + o; o += N_ASSOC;
    float* in_spk= ws + o; o += (size_t)T_STEPS * N_VIS;
    float* Wva   = ws + o; o += (size_t)N_ASSOC * N_VIS;   // mutable copy
    float* Wam   = ws + o; o += (size_t)N_MOTOR * N_ASSOC; // mutable copy

    // Fresh state every call (deterministic across graph replays).
    hipMemsetAsync(ws, 0, zero_floats * sizeof(float), stream);
    hipMemcpyAsync(Wva, Wva_in, (size_t)N_ASSOC * N_VIS * sizeof(float),
                   hipMemcpyDeviceToDevice, stream);
    hipMemcpyAsync(Wam, Wam_in, (size_t)N_MOTOR * N_ASSOC * sizeof(float),
                   hipMemcpyDeviceToDevice, stream);

    // 1) Hoisted encoder GEMM for all 16 timesteps (512 tiles, 8 waves/block).
    snn_encoder_wmma<<<(N_VIS / 16) / 8, 256, 0, stream>>>(
        visual, Wenc, benc, noise, in_spk);

    // 2) Sequential 16-step scan.
    for (int t = 0; t < T_STEPS; ++t) {
        snn_vis_step<<<N_VIS / 256, 256, 0, stream>>>(vV, in_spk + (size_t)t * N_VIS, v_spk);

        snn_lif_matvec<<<N_ASSOC / 8, 256, 0, stream>>>(Wva, v_spk, vA, a_spk, N_VIS);
        snn_lif_matvec<<<N_MOTOR / 8, 256, 0, stream>>>(Wam, a_spk, vM,
                                                        out + (size_t)t * N_MOTOR, N_ASSOC);

        snn_elig_update<<<((size_t)N_ASSOC * N_VIS / 4) / 256, 256, 0, stream>>>(
            E_va, Wva, a_spk, v_spk, dopa, t, N_VIS);
        snn_elig_update<<<((size_t)N_MOTOR * N_ASSOC / 4) / 256, 256, 0, stream>>>(
            E_am, Wam, out + (size_t)t * N_MOTOR, a_spk, dopa, t, N_ASSOC);
    }
}
